// NeuralParametrizedBSDF3_76828374991340
// MI455X (gfx1250) — matrix operand order repaired and checked by reading, not compiled, not run on hardware
//
#include <hip/hip_runtime.h>
#include <math.h>

typedef __attribute__((ext_vector_type(16))) _Float16 v16h;
typedef __attribute__((ext_vector_type(8)))  float    v8f;
typedef __attribute__((ext_vector_type(4)))  unsigned int v4u;

union Frag16 { v16h h; v4u u[2]; };

#define BLOCKT 256
#define NWAVES 8
#define ROWS_PER_WAVE 32
#define ROWS_PER_CHUNK 256   // NWAVES * ROWS_PER_WAVE
#define MAX_BLOCKS 1024
#define PI_F 3.14159265358979323846f

// ---- LDS layout (offsets in halves for the f16 region) ----
static constexpr int OFF_W1    = 0;                 // 64 x 32
static constexpr int OFF_WX[8] = {2048, 6144, 10240, 14336, 20480, 24576, 28672, 34816};
static constexpr int KPAD_X[8] = {64, 64, 64, 96, 64, 64, 96, 64};
static constexpr int OFF_WFEAT = 40960;             // 64 x 64
static constexpr int OFF_WS0   = 45056;             // 32 x 96
static constexpr int OFF_WS1   = 48128;             // 32 x 32
static constexpr int OFF_ORIG  = 49152;             // 8 waves * 32 rows * 32
static constexpr int OFF_ACT   = 57344;             // 8 waves * 32 rows * 96
static constexpr int HALVES_TOTAL = 81920;
// ---- f32 region offsets (in floats, after the halves) ----
static constexpr int FB1     = 0;     // 64
static constexpr int FBX     = 64;    // 8*64
static constexpr int FBFEAT  = 576;   // 64
static constexpr int FBS0    = 640;   // 32
static constexpr int FBS1    = 672;   // 32
static constexpr int FWALPHA = 704;   // 64
static constexpr int FBALPHA = 768;   // 1
static constexpr int FWRGB   = 769;   // 96
static constexpr int FBRGB   = 865;   // 3
static constexpr int FALPHA  = 868;   // 256
static constexpr int FLOATS_TOTAL = 1124;
static constexpr size_t SMEM_BYTES = (size_t)HALVES_TOTAL*2 + (size_t)FLOATS_TOTAL*4;

struct Params {
  const float *x, *basis_p, *basis_ang;
  const float *W1, *b1;
  const float *xw[8];
  const float *xb[8];
  const float *Wfeat, *bfeat, *Walpha, *balpha;
  const float *Ws0, *bs0, *Ws1, *bs1, *Wrgb, *brgb;
  float* out;
  int N;
};

struct F3 { float x, y, z; };

__device__ inline F3 nrm3(F3 v) {
  float n = sqrtf(v.x*v.x + v.y*v.y + v.z*v.z);
  float inv = 1.0f / fmaxf(n, 1e-12f);
  return F3{v.x*inv, v.y*inv, v.z*inv};
}
__device__ inline F3 rot3(F3 v, F3 a, float c, float s) {
  float d = v.x*a.x + v.y*a.y + v.z*a.z;
  float k = d * (1.0f - c);
  F3 cr{ a.y*v.z - a.z*v.y, a.z*v.x - a.x*v.z, a.x*v.y - a.y*v.x };
  return F3{ v.x*c + a.x*k + cr.x*s,
             v.y*c + a.y*k + cr.y*s,
             v.z*c + a.z*k + cr.z*s };
}

// stage W (out x in, row-major f32) into LDS as f16 row-major with K padded to kpad
__device__ inline void stage_w(const float* W, _Float16* dst, int out, int in, int kpad) {
  for (int idx = threadIdx.x; idx < out * kpad; idx += BLOCKT) {
    int n = idx / kpad, k = idx - n * kpad;
    dst[idx] = (_Float16)((k < in) ? W[n * in + k] : 0.0f);
  }
}
__device__ inline void stage_f(const float* src, float* dst, int cnt) {
  for (int idx = threadIdx.x; idx < cnt; idx += BLOCKT) dst[idx] = src[idx];
}

// Dense layer via WMMA over TWO 16-row tiles per wave (B fragments reused):
// D(32 x NT*16) = A(32 x KT*32) * B + bias (+ReLU)
template <int KT, int NT>
__device__ inline void mlp_layer(const _Float16* aBase, int aStride,
                                 const _Float16* wB, int kpad,
                                 const float* bias, bool doRelu,
                                 _Float16* dBase, int dStride, int lane) {
  const int m   = lane & 15;
  const int kbA = (lane & 16) ? 8 : 0;    // A: K-half offset (halves)
  const int kbB = (lane & 16) ? 8 : 0;    // B: K-half offset (halves)
  const int mo  = (lane & 16) ? 8 : 0;    // C/D: row offset
  Frag16 A0[KT], A1[KT];
#pragma unroll
  for (int kt = 0; kt < KT; ++kt) {
    const _Float16* p0 = aBase + m * aStride + kt * 32 + kbA;
    A0[kt].u[0] = *(const v4u*)(const void*)p0;
    A0[kt].u[1] = *(const v4u*)(const void*)(p0 + 16);
    const _Float16* p1 = p0 + 16 * aStride;
    A1[kt].u[0] = *(const v4u*)(const void*)p1;
    A1[kt].u[1] = *(const v4u*)(const void*)(p1 + 16);
  }
#pragma unroll
  for (int nt = 0; nt < NT; ++nt) {
    v8f acc0 = {0.f,0.f,0.f,0.f,0.f,0.f,0.f,0.f};
    v8f acc1 = {0.f,0.f,0.f,0.f,0.f,0.f,0.f,0.f};
#pragma unroll
    for (int kt = 0; kt < KT; ++kt) {
      Frag16 B;
      const _Float16* p = wB + (nt * 16 + m) * kpad + kt * 32 + kbB;
      B.u[0] = *(const v4u*)(const void*)p;
      B.u[1] = *(const v4u*)(const void*)(p + 16);
      acc0 = __builtin_amdgcn_wmma_f32_16x16x32_f16(false, A0[kt].h, false, B.h,
                                                    (short)0, acc0, false, false);
      acc1 = __builtin_amdgcn_wmma_f32_16x16x32_f16(false, A1[kt].h, false, B.h,
                                                    (short)0, acc1, false, false);
    }
    float bv = bias[nt * 16 + m];
#pragma unroll
    for (int v = 0; v < 8; ++v) {
      float v0 = acc0[v] + bv;
      float v1 = acc1[v] + bv;
      if (doRelu) { v0 = fmaxf(v0, 0.0f); v1 = fmaxf(v1, 0.0f); }
      dBase[(mo + v) * dStride + nt * 16 + m]        = (_Float16)v0;
      dBase[(16 + mo + v) * dStride + nt * 16 + m]   = (_Float16)v1;
    }
  }
}

__global__ void __launch_bounds__(BLOCKT)
bsdf_mlp_kernel(Params P) {
  extern __shared__ __align__(16) char smem[];
  _Float16* sH = (_Float16*)smem;
  float*    sF = (float*)(smem + (size_t)HALVES_TOTAL * 2);

  const int tid  = threadIdx.x;
  const int lane = tid & 31;
  const int wave = tid >> 5;

  // ---------------- Phase 0: stage weights/biases into LDS (once) ---------
  stage_w(P.W1, sH + OFF_W1, 64, 22, 32);
#pragma unroll
  for (int i = 0; i < 8; ++i) {
    int in_d = (i == 3 || i == 6) ? 86 : 64;
    stage_w(P.xw[i], sH + OFF_WX[i], 64, in_d, KPAD_X[i]);
    stage_f(P.xb[i], sF + FBX + i * 64, 64);
  }
  stage_w(P.Wfeat, sH + OFF_WFEAT, 64, 64, 64);
  stage_w(P.Ws0,   sH + OFF_WS0,   32, 86, 96);
  stage_w(P.Ws1,   sH + OFF_WS1,   32, 32, 32);
  stage_f(P.b1,     sF + FB1, 64);
  stage_f(P.bfeat,  sF + FBFEAT, 64);
  stage_f(P.bs0,    sF + FBS0, 32);
  stage_f(P.bs1,    sF + FBS1, 32);
  stage_f(P.Walpha, sF + FWALPHA, 64);
  stage_f(P.balpha, sF + FBALPHA, 1);
  stage_f(P.Wrgb,   sF + FWRGB, 96);
  stage_f(P.brgb,   sF + FBRGB, 3);

  _Float16*       act = sH + OFF_ACT  + wave * (ROWS_PER_WAVE * 96);
  const _Float16* org = sH + OFF_ORIG + wave * (ROWS_PER_WAVE * 32);

  const int numChunks = (P.N + ROWS_PER_CHUNK - 1) / ROWS_PER_CHUNK;

  // ------------- Persistent loop: 256 rows per chunk per block ------------
  for (int chunk = blockIdx.x; chunk < numChunks; chunk += gridDim.x) {
    const int rowBase = chunk * ROWS_PER_CHUNK;

    // -------- Phase 1: per-point preprocessing (f32 VALU), 1 row/thread ---
    {
      int row = rowBase + tid;
      _Float16* o = sH + OFF_ORIG + tid * 32;
      if (row < P.N) {
        const float* xr = P.x + (size_t)row * 14;
        float p0 = xr[2], p1 = xr[3], p2 = xr[4];
        F3 n   = {xr[5], xr[6], xr[7]};
        F3 woI = {xr[8], xr[9], xr[10]};
        F3 wiI = {xr[11], xr[12], xr[13]};
        // reference calls _parametrize(n, wi, wo): wo := wi_input, wi := wo_input
        F3 wo = nrm3(wiI);
        F3 wi = nrm3(woI);
        n = nrm3(n);
        bool mask = (n.x != 0.f) || (n.y != 0.f) || (n.z != 0.f);
        F3 mid = nrm3(F3{n.x * 0.5f, n.y * 0.5f, (n.z + 1.0f) * 0.5f});
        if (mask) {
          n  = F3{0.f, 0.f, 1.f};
          wo = rot3(wo, mid, -1.0f, 0.0f);
          wi = rot3(wi, mid, -1.0f, 0.0f);
        } else {
          wo = F3{0.f, 0.f, 0.f};
          wi = F3{0.f, 0.f, 0.f};
        }
        F3 Hv = nrm3(F3{(wo.x + wi.x) * 0.5f, (wo.y + wi.y) * 0.5f, (wo.z + wi.z) * 0.5f});
        float theta_h = acosf(fminf(fmaxf(Hv.z, -1.0f), 1.0f));
        float phi_h   = atan2f(Hv.y, Hv.x);
        float cv = cosf(-phi_h), sv = sinf(-phi_h);
        F3 tmp = nrm3(rot3(wi, n, cv, sv));
        cv = cosf(-theta_h); sv = sinf(-theta_h);
        F3 diff = nrm3(rot3(tmp, F3{0.f, 1.f, 0.f}, cv, sv));
        float theta_d = acosf(fminf(fmaxf(diff.z, 0.0f), 1.0f));
        float phi_d   = fmodf(atan2f(diff.y, diff.x), PI_F);
        float a0 = mask ? phi_d   : 0.f;
        float a1 = mask ? theta_h : 0.f;
        float a2 = mask ? theta_d : 0.f;
        o[0] = (_Float16)p0; o[1] = (_Float16)p1; o[2] = (_Float16)p2;
#pragma unroll
        for (int j = 0; j < 4; ++j) {
          float inner = p0 * P.basis_p[j*3] + p1 * P.basis_p[j*3+1] + p2 * P.basis_p[j*3+2];
          o[3 + 2*j] = (_Float16)sinf(inner);
          o[4 + 2*j] = (_Float16)cosf(inner);
        }
        o[11] = (_Float16)a0; o[12] = (_Float16)a1; o[13] = (_Float16)a2;
#pragma unroll
        for (int j = 0; j < 4; ++j) {
          float inner = a0 * P.basis_ang[j*3] + a1 * P.basis_ang[j*3+1] + a2 * P.basis_ang[j*3+2];
          o[14 + 2*j] = (_Float16)sinf(inner);
          o[15 + 2*j] = (_Float16)cosf(inner);
        }
        for (int c = 22; c < 32; ++c) o[c] = (_Float16)0.f;
      } else {
        for (int c = 0; c < 32; ++c) o[c] = (_Float16)0.f;
      }
    }
    __syncthreads();

    // -------- Phase 2: WMMA MLP, one 32-row tile per wave -----------------
    // h = orig @ W1^T + b1   (22->64, no activation)
    mlp_layer<1, 4>(org, 32, sH + OFF_W1, 32, sF + FB1, false, act, 96, lane);
    __syncthreads();

#pragma unroll
    for (int i = 0; i < 8; ++i) {
      if (i == 3 || i == 6) {
        // concat orig into act columns 64..95 (cols 86..95 are zero padding)
        for (int idx = lane; idx < ROWS_PER_WAVE * 32; idx += 32) {
          int r = idx >> 5, c = idx & 31;
          act[r * 96 + 64 + c] = org[r * 32 + c];
        }
        __syncthreads();
        mlp_layer<3, 4>(act, 96, sH + OFF_WX[i], 96, sF + FBX + i * 64, true, act, 96, lane);
      } else {
        mlp_layer<2, 4>(act, 96, sH + OFF_WX[i], 64, sF + FBX + i * 64, true, act, 96, lane);
      }
      __syncthreads();
    }

    // alpha = sigmoid(h @ Walpha^T + balpha) -- all 32 lanes, 1 row each
    {
      float a = sF[FBALPHA];
      for (int k = 0; k < 64; ++k) a += (float)act[lane * 96 + k] * sF[FWALPHA + k];
      sF[FALPHA + wave * ROWS_PER_WAVE + lane] = 1.0f / (1.0f + expf(-a));
    }
    __syncthreads();

    // feat = relu(h @ Wfeat^T + bfeat) -> act cols 0..63 (orig still at 64..85)
    mlp_layer<2, 4>(act, 96, sH + OFF_WFEAT, 64, sF + FBFEAT, true, act, 96, lane);
    __syncthreads();
    // s = relu(concat(feat, orig) @ Ws0^T + bs0)  (86->32) -> act cols 0..31
    mlp_layer<3, 2>(act, 96, sH + OFF_WS0, 96, sF + FBS0, true, act, 96, lane);
    __syncthreads();
    // s = relu(s @ Ws1^T + bs1) (32->32)
    mlp_layer<1, 2>(act, 96, sH + OFF_WS1, 32, sF + FBS1, true, act, 96, lane);
    __syncthreads();

    // rgb = s @ Wrgb^T + brgb ; out = rgb * alpha  -- all 32 lanes, 1 row each
    {
      int row = rowBase + wave * ROWS_PER_WAVE + lane;
      if (row < P.N) {
        float al = sF[FALPHA + wave * ROWS_PER_WAVE + lane];
#pragma unroll
        for (int c = 0; c < 3; ++c) {
          float r2 = sF[FBRGB + c];
          for (int k = 0; k < 32; ++k) r2 += (float)act[lane * 96 + k] * sF[FWRGB + c * 32 + k];
          P.out[(size_t)row * 3 + c] = r2 * al;
        }
      }
    }
    __syncthreads();  // protect sOrig/sAct before next chunk overwrites
  }
}

extern "C" void kernel_launch(void* const* d_in, const int* in_sizes, int n_in,
                              void* d_out, int out_size, void* d_ws, size_t ws_size,
                              hipStream_t stream) {
  (void)d_ws; (void)ws_size; (void)out_size;
  Params P;
  P.x         = (const float*)d_in[0];
  P.basis_p   = (const float*)d_in[1];
  P.basis_ang = (const float*)d_in[2];
  P.W1        = (const float*)d_in[3];
  P.b1        = (const float*)d_in[4];
  if (n_in >= 31) {
    // lists flattened: xyz_w[0..7] at 5..12, xyz_b[0..7] at 13..20
    for (int i = 0; i < 8; ++i) { P.xw[i] = (const float*)d_in[5 + i]; P.xb[i] = (const float*)d_in[13 + i]; }
    P.Wfeat  = (const float*)d_in[21]; P.bfeat  = (const float*)d_in[22];
    P.Walpha = (const float*)d_in[23]; P.balpha = (const float*)d_in[24];
    P.Ws0    = (const float*)d_in[25]; P.bs0    = (const float*)d_in[26];
    P.Ws1    = (const float*)d_in[27]; P.bs1    = (const float*)d_in[28];
    P.Wrgb   = (const float*)d_in[29]; P.brgb   = (const float*)d_in[30];
  } else {
    // fallback: lists passed as single concatenated blobs at index 5 / 6
    const float* wb = (const float*)d_in[5];
    size_t o = 0;
    for (int i = 0; i < 8; ++i) {
      P.xw[i] = wb + o;
      o += (size_t)64 * ((i == 3 || i == 6) ? 86 : 64);
    }
    const float* bb = (const float*)d_in[6];
    for (int i = 0; i < 8; ++i) P.xb[i] = bb + (size_t)i * 64;
    P.Wfeat  = (const float*)d_in[7];  P.bfeat  = (const float*)d_in[8];
    P.Walpha = (const float*)d_in[9];  P.balpha = (const float*)d_in[10];
    P.Ws0    = (const float*)d_in[11]; P.bs0    = (const float*)d_in[12];
    P.Ws1    = (const float*)d_in[13]; P.bs1    = (const float*)d_in[14];
    P.Wrgb   = (const float*)d_in[15]; P.brgb   = (const float*)d_in[16];
  }
  P.out = (float*)d_out;
  P.N   = in_sizes[0] / 14;

  (void)hipFuncSetAttribute((const void*)bsdf_mlp_kernel,
                            hipFuncAttributeMaxDynamicSharedMemorySize,
                            (int)SMEM_BYTES);
  int numChunks = (P.N + ROWS_PER_CHUNK - 1) / ROWS_PER_CHUNK;
  int blocks = numChunks < MAX_BLOCKS ? numChunks : MAX_BLOCKS;
  if (blocks < 1) blocks = 1;
  hipLaunchKernelGGL(bsdf_mlp_kernel, dim3(blocks), dim3(BLOCKT), SMEM_BYTES, stream, P);
}
